// Conv2X1_36971078484648
// MI455X (gfx1250) — matrix operand order repaired and checked by reading, not compiled, not run on hardware
//
#include <hip/hip_runtime.h>

// ---------------- problem constants (from reference) ----------------
#define CIN   512
#define COUT  512
#define TT    4096
#define NBAT  16
// ---------------- tiling ----------------
#define BM 256
#define BN 128
#define BK 32
#define NTHREADS 512        // 16 waves: 4 (M) x 4 (N), wave tile 64x32
#define KSTEPS (CIN / BK)   // 16
#define LDA 40              // padded A row stride in bf16 elements (32 + 8)
#define LDB 40              // padded B column stride in bf16 elements (32 + 8)

typedef __attribute__((ext_vector_type(16))) __bf16 bf16x16;
typedef __attribute__((ext_vector_type(8)))  float  f32x8;

// Single-instruction f32x2 -> packed bf16 (RNE). The compiler already emits
// v_cvt_pk_bf16_f32 on gfx1250; forcing it avoids SLP repack shuffles.
__device__ __forceinline__ unsigned pack2bf(float a, float b) {
  unsigned r;
  asm("v_cvt_pk_bf16_f32 %0, %1, %2" : "=v"(r) : "v"(a), "v"(b));
  return r;
}

// One-shot: W (f32, [COUT,CIN]) -> bf16 into workspace; bias2 = 2*b (f32).
__global__ void convert_w_bias(const float* __restrict__ W,
                               const float* __restrict__ b,
                               unsigned short* __restrict__ Wbf,
                               float* __restrict__ b2) {
  int i = blockIdx.x * 256 + threadIdx.x;
  if (i < COUT * CIN) Wbf[i] = __builtin_bit_cast(unsigned short, (__bf16)W[i]);
  if (i < COUT)       b2[i]  = 2.0f * b[i];
}

// D[b] = 2*(W @ pre[b]) + 2*bias  via v_wmma_f32_16x16x32_bf16
__launch_bounds__(NTHREADS)
__global__ void gemm_bf16_wmma(const float* __restrict__ pre,
                               const unsigned short* __restrict__ Wbf,
                               const float* __restrict__ bias2,
                               float* __restrict__ out) {
  // A tile: BM rows x BK bf16, row-major, padded. B tile: BN cols x BK bf16, K-major, padded.
  __shared__ alignas(16) unsigned short Abuf[2][BM * LDA];  // 40 KB
  __shared__ alignas(16) unsigned short Bbuf[2][BN * LDB];  // 20 KB

  const int tid  = threadIdx.x;
  const int lane = tid & 31;
  const int wv   = tid >> 5;     // wave id 0..15
  const int wm   = wv & 3;       // wave M index (0..3) -> 64-row slab
  const int wn   = wv >> 2;      // wave N index (0..3) -> 32-col slab
  const int lhi  = lane >> 4;    // K-half selector per WMMA operand layout
  const int llo  = lane & 15;

  const int N0  = blockIdx.x * BN;
  const int M0  = blockIdx.y * BM;
  const int bat = blockIdx.z;

  const float* preB = pre + (size_t)bat * CIN * TT;
  const unsigned long long wbase = (unsigned long long)(size_t)Wbf;

  f32x8 acc[4][2];
#pragma unroll
  for (int i = 0; i < 4; ++i)
#pragma unroll
    for (int j = 0; j < 2; ++j)
#pragma unroll
      for (int k = 0; k < 8; ++k) acc[i][j][k] = 0.0f;

  // A staging: 256 rows x 64B; each thread DMAs two 16B chunks (async to LDS).
  const int ar0 = tid >> 2;   // row 0..127 (and +128)
  const int ac4 = tid & 3;    // which 16B chunk of the 64B row
  // B staging: wave wv owns c-rows {2wv, 2wv+1}; lanes sweep t in float4s.
  const int tloc = lane * 4;
  const int c0   = wv * 2;

  float4 ba, bb;

  // Async DMA of the A (W, already bf16) tile straight into LDS. GVS mode:
  // mem = SGPR base + VGPR byte-offset. VDST = 32-bit LDS byte address.
  auto asyncLoadA = [&](int buf, int k0) {
    unsigned lds0 = (unsigned)(size_t)&Abuf[buf][(size_t)ar0         * LDA + ac4 * 8];
    unsigned lds1 = (unsigned)(size_t)&Abuf[buf][(size_t)(ar0 + 128) * LDA + ac4 * 8];
    unsigned off0 = (unsigned)(((M0 + ar0)       * CIN + k0 + ac4 * 8) * 2);
    unsigned off1 = (unsigned)(((M0 + 128 + ar0) * CIN + k0 + ac4 * 8) * 2);
    asm volatile("global_load_async_to_lds_b128 %0, %1, %4\n\t"
                 "global_load_async_to_lds_b128 %2, %3, %4"
                 :: "v"(lds0), "v"(off0), "v"(lds1), "v"(off1), "s"(wbase)
                 : "memory");
  };
  auto waitAsync = [&]() {
    asm volatile("s_wait_asynccnt 0x0" ::: "memory");
  };

  auto loadB = [&](int k0) {
    ba = *(const float4*)(preB + (size_t)(k0 + c0)     * TT + N0 + tloc);
    bb = *(const float4*)(preB + (size_t)(k0 + c0 + 1) * TT + N0 + tloc);
  };
  auto storeB = [&](int buf) {
    float fa[4] = {ba.x, ba.y, ba.z, ba.w};
    float fb[4] = {bb.x, bb.y, bb.z, bb.w};
#pragma unroll
    for (int e = 0; e < 4; ++e)
      *(unsigned*)&Bbuf[buf][(size_t)(tloc + e) * LDB + c0] = pack2bf(fa[e], fb[e]);
  };

  asyncLoadA(0, 0);
  loadB(0); storeB(0);
  waitAsync();
  __syncthreads();

#pragma unroll 2
  for (int ks = 0; ks < KSTEPS; ++ks) {
    const int cur = ks & 1;
    if (ks + 1 < KSTEPS) {
      asyncLoadA(cur ^ 1, (ks + 1) * BK);   // DMA next A tile while we compute
      loadB((ks + 1) * BK);                 // prefetch next B tile into regs
    }

    union Frag { uint4 u[2]; bf16x16 v; };
    Frag af[4], bfr[2];
#pragma unroll
    for (int i = 0; i < 4; ++i) {
      const int row = wm * 64 + i * 16 + llo;
      const unsigned short* p = &Abuf[cur][(size_t)row * LDA + lhi * 8];
      af[i].u[0] = *(const uint4*)(p);        // K = off .. off+7
      af[i].u[1] = *(const uint4*)(p + 16);   // K = off+16 .. off+23
    }
#pragma unroll
    for (int j = 0; j < 2; ++j) {
      const int col = wn * 32 + j * 16 + llo;
      const unsigned short* p = &Bbuf[cur][(size_t)col * LDB + lhi * 8];
      bfr[j].u[0] = *(const uint4*)(p);       // K = half*8 + 0..7
      bfr[j].u[1] = *(const uint4*)(p + 16);  // K = 16 + half*8 + 0..7
    }

#pragma unroll
    for (int i = 0; i < 4; ++i)
#pragma unroll
      for (int j = 0; j < 2; ++j)
        acc[i][j] = __builtin_amdgcn_wmma_f32_16x16x32_bf16(
            false, af[i].v, false, bfr[j].v, (short)0, acc[i][j], false, false);

    if (ks + 1 < KSTEPS) {
      storeB(cur ^ 1);
      waitAsync();                            // A-tile DMA for next step complete
    }
    __syncthreads();
  }

  // Epilogue: out = 2*acc + 2*bias  (bias2 already holds 2*b).
#pragma unroll
  for (int i = 0; i < 4; ++i) {
    const int mbase = M0 + wm * 64 + i * 16 + 8 * lhi;
#pragma unroll
    for (int v = 0; v < 8; ++v) {
      const int m  = mbase + v;
      const float bv = bias2[m];
#pragma unroll
      for (int j = 0; j < 2; ++j) {
        const int n = N0 + wn * 32 + j * 16 + llo;
        out[((size_t)bat * COUT + m) * TT + n] = fmaf(2.0f, acc[i][j][v], bv);
      }
    }
  }
}

extern "C" void kernel_launch(void* const* d_in, const int* in_sizes, int n_in,
                              void* d_out, int out_size, void* d_ws, size_t ws_size,
                              hipStream_t stream) {
  const float* pre  = (const float*)d_in[0];  // [16, 512, 4096] f32
  const float* W    = (const float*)d_in[1];  // [512, 512] f32
  const float* bias = (const float*)d_in[2];  // [512] f32
  float* out = (float*)d_out;                 // [16, 512, 4096] f32

  unsigned short* Wbf = (unsigned short*)d_ws;                         // 512 KB bf16 W
  float* b2 = (float*)((char*)d_ws + (size_t)COUT * CIN * sizeof(unsigned short));

  convert_w_bias<<<(COUT * CIN + 255) / 256, 256, 0, stream>>>(W, bias, Wbf, b2);

  dim3 grid(TT / BN, COUT / BM, NBAT);   // 32 x 2 x 16 = 1024 blocks
  gemm_bf16_wmma<<<grid, NTHREADS, 0, stream>>>(pre, Wbf, b2, out);
}
